// TimeAwareNodeModel_49976239456338
// MI455X (gfx1250) — matrix operand order repaired and checked by reading, not compiled, not run on hardware
//
#include <hip/hip_runtime.h>
#include <hip/hip_bf16.h>

// ---------------------------------------------------------------------------
// TimeAwareNodeModel for MI455X (gfx1250): bf16 WMMA edge/node MLPs.
//   N=50000 nodes, E=1600000 edges, d_node=32, d_edge=16, H=64, F_OUT=32.
// Edge MLP K-dim = 80 -> padded to 96 (3 K-tiles of 32).
// Aggregation buffer agg[N][64] f32 in d_ws (fwd cols 0..31, bwd cols 32..63).
// Streaming inputs (edge_attr, edge_index) use non-temporal loads so L2 keeps
// the hot working set (x: 6.4MB, agg: 12.8MB) resident for gathers/atomics.
// ---------------------------------------------------------------------------

#define NN      50000
#define NE      1600000
#define AGG_F32 (NN * 64)                  // 3,200,000 floats = 12.8 MB
#define AGG_BYTES (AGG_F32 * 4)
#define N_FRAG  44                         // 12+12+4+4 edge frags, 8+4 node frags
#define FRAG_U32 (N_FRAG * 256)            // each frag: 32 lanes x 8 dwords
#define BIAS_F32 288

typedef __attribute__((ext_vector_type(16))) __bf16 v16bf;
typedef __attribute__((ext_vector_type(2)))  __bf16 bf2;
typedef __attribute__((ext_vector_type(8)))  float  v8f;
typedef __attribute__((ext_vector_type(2)))  float  f2;

static __device__ __forceinline__ v8f wmma_bf16(v16bf a, v16bf b, v8f c) {
  return __builtin_amdgcn_wmma_f32_16x16x32_bf16(false, a, false, b, (short)0, c,
                                                 false, false);
}

// Pack two f32 -> one dword of two bf16 (should lower to v_cvt_pk_bf16_f32).
static __device__ __forceinline__ unsigned pack2(float a, float b) {
  bf2 t;
  t.x = (__bf16)a;
  t.y = (__bf16)b;
  return __builtin_bit_cast(unsigned, t);
}

union AFrag { unsigned u[8]; v16bf v; };

static __device__ __forceinline__ v16bf load_frag(const unsigned* p) {
  union { uint4 q[2]; v16bf v; } u;
  u.q[0] = *(const uint4*)(p);
  u.q[1] = *(const uint4*)(p + 4);
  return u.v;
}

// A-fragment K index for 16-bit A (16xK) per CDNA5 ISA layout:
// lanes 0-15: VGPR0..3 -> K 0..7, VGPR4..7 -> K 16..23; lanes 16-31: +8.
static __device__ __forceinline__ int a_k(int j, int half) {
  return ((j >> 2) << 4) + (half << 3) + ((j & 3) << 1);
}

// ---------------------------------------------------------------------------
// Setup: repack f32 weights into bf16 WMMA B-fragments + gather biases.
// Frag table layout (frag index):
//   [ 0,12) W1f  [kt:3][nt:4]   K padded 80->96, N=64
//   [12,24) W1b
//   [24,28) W2f  [kt:2][nt:2]   K=64, N=32
//   [28,32) W2b
//   [32,40) W1n  [kt:2][nt:4]   K=64, N=64
//   [40,44) W2n  [kt:2][nt:2]   K=64, N=32
// B element pair (2j,2j+1) of lane L  <->  k = kt*32 + (j/4)*16 + (L/16)*8 + 2*(j%4), n = nt*16 + L%16
// ---------------------------------------------------------------------------
__global__ __launch_bounds__(256) void gnn_pack_kernel(
    const float* __restrict__ W1f, const float* __restrict__ W2f,
    const float* __restrict__ W1b, const float* __restrict__ W2b,
    const float* __restrict__ W1n, const float* __restrict__ W2n,
    const float* __restrict__ b1f, const float* __restrict__ b2f,
    const float* __restrict__ b1b, const float* __restrict__ b2b,
    const float* __restrict__ b1n, const float* __restrict__ b2n,
    unsigned* __restrict__ frags, float* __restrict__ biasbuf) {
  const int t = blockIdx.x * 256 + threadIdx.x;
  if (t < FRAG_U32) {
    const int f = t >> 8, r = t & 255;
    const int lane = r >> 3, j = r & 7;
    const int half = lane >> 4, nI = lane & 15;
    const int kl = ((j >> 2) << 4) + half * 8 + ((j & 3) << 1);
    const float* W; int kt, nt, Nc, Kv;
    if (f < 12)      { W = W1f; int g = f;      kt = g >> 2; nt = g & 3; Nc = 64; Kv = 80; }
    else if (f < 24) { W = W1b; int g = f - 12; kt = g >> 2; nt = g & 3; Nc = 64; Kv = 80; }
    else if (f < 28) { W = W2f; int g = f - 24; kt = g >> 1; nt = g & 1; Nc = 32; Kv = 64; }
    else if (f < 32) { W = W2b; int g = f - 28; kt = g >> 1; nt = g & 1; Nc = 32; Kv = 64; }
    else if (f < 40) { W = W1n; int g = f - 32; kt = g >> 2; nt = g & 3; Nc = 64; Kv = 64; }
    else             { W = W2n; int g = f - 40; kt = g >> 1; nt = g & 1; Nc = 32; Kv = 64; }
    const int k0 = kt * 32 + kl;
    const int n  = nt * 16 + nI;
    const float v0 = (k0     < Kv) ? W[k0 * Nc + n]       : 0.f;
    const float v1 = (k0 + 1 < Kv) ? W[(k0 + 1) * Nc + n] : 0.f;
    frags[t] = pack2(v0, v1);
  } else {
    const int i = t - FRAG_U32;
    if (i < BIAS_F32) {
      float v;
      if (i < 64)       v = b1f[i];
      else if (i < 96)  v = b2f[i - 64];
      else if (i < 160) v = b1b[i - 96];
      else if (i < 192) v = b2b[i - 160];
      else if (i < 256) v = b1n[i - 192];
      else              v = b2n[i - 256];
      biasbuf[i] = v;
    }
  }
}

__global__ __launch_bounds__(256) void gnn_zero_kernel(float4* __restrict__ p, int n4) {
  int i = blockIdx.x * 256 + threadIdx.x;
  const int stride = gridDim.x * 256;
  const float4 z = make_float4(0.f, 0.f, 0.f, 0.f);
  for (; i < n4; i += stride) p[i] = z;
}

// ---------------------------------------------------------------------------
// Node kernel: relu(relu(agg[N,64] @ W1n + b1n) @ W2n + b2n) -> out[N,32]
// (Defined before the edge kernel so its WMMA block appears early in disasm.)
// ---------------------------------------------------------------------------
__global__ __launch_bounds__(256) void gnn_node_kernel(
    const float* __restrict__ agg, const unsigned* __restrict__ gfrags,
    const float* __restrict__ gbias, float* __restrict__ out, int numTiles) {
  __shared__ unsigned sfr[12 * 256];        // W1n (8) + W2n (4) fragments
  __shared__ float    sb[96];               // b1n, b2n
  __shared__ __bf16   sstage[8][1024];

  for (int i = threadIdx.x; i < (12 * 256) / 4; i += 256)
    ((uint4*)sfr)[i] = ((const uint4*)gfrags)[i];
  if (threadIdx.x < 96) sb[threadIdx.x] = gbias[threadIdx.x];
  __syncthreads();

  const int lane = threadIdx.x & 31;
  const int wv   = threadIdx.x >> 5;
  const int half = lane >> 4;
  const int lm   = lane & 15;
  __bf16* stage  = sstage[wv];

  for (int tile = blockIdx.x * 8 + wv; tile < numTiles; tile += gridDim.x * 8) {
    const int nb  = tile << 4;
    const int row = nb + lm;

    AFrag g0, g1;
#pragma unroll
    for (int j = 0; j < 8; ++j) {
      const int k = a_k(j, half);
      const f2 p = *(const f2*)&agg[row * 64 + k];
      g0.u[j] = pack2(p.x, p.y);
      const f2 q = *(const f2*)&agg[row * 64 + 32 + k];
      g1.u[j] = pack2(q.x, q.y);
    }

#pragma unroll
    for (int nt = 0; nt < 4; ++nt) {
      v8f c = {};
      c = wmma_bf16(g0.v, load_frag(&sfr[(0 * 4 + nt) * 256] + lane * 8), c);
      c = wmma_bf16(g1.v, load_frag(&sfr[(1 * 4 + nt) * 256] + lane * 8), c);
      const float bb = sb[nt * 16 + lm];
#pragma unroll
      for (int v = 0; v < 8; ++v) {
        float h = c[v] + bb;
        h = h > 0.f ? h : 0.f;
        stage[(v + 8 * half) * 64 + nt * 16 + lm] = (__bf16)h;
      }
    }

    AFrag h0, h1;
#pragma unroll
    for (int j = 0; j < 8; ++j) {
      const int kk = a_k(j, half);
      h0.u[j] = *(const unsigned*)&stage[lm * 64 + kk];
      h1.u[j] = *(const unsigned*)&stage[lm * 64 + 32 + kk];
    }

#pragma unroll
    for (int nt = 0; nt < 2; ++nt) {
      v8f c = {};
      c = wmma_bf16(h0.v, load_frag(&sfr[(8 + 0 * 2 + nt) * 256] + lane * 8), c);
      c = wmma_bf16(h1.v, load_frag(&sfr[(8 + 1 * 2 + nt) * 256] + lane * 8), c);
      const float bb = sb[64 + nt * 16 + lm];
#pragma unroll
      for (int v = 0; v < 8; ++v) {
        float val = c[v] + bb;
        val = val > 0.f ? val : 0.f;
        // Write-once output: non-temporal store.
        __builtin_nontemporal_store(val, &out[(nb + v + 8 * half) * 32 + nt * 16 + lm]);
      }
    }
  }
}

// ---------------------------------------------------------------------------
// Edge kernel: one wave per 16-edge tile, both directions per tile.
// ---------------------------------------------------------------------------
__global__ __launch_bounds__(256) void gnn_edge_kernel(
    const float* __restrict__ x, const int* __restrict__ eidx,
    const float* __restrict__ eattr, const unsigned* __restrict__ gfrags,
    const float* __restrict__ gbias, float* __restrict__ agg, int numTiles) {
  __shared__ unsigned sfr[32 * 256];        // W1f/W1b/W2f/W2b fragments (32 KB)
  __shared__ float    sb[192];              // b1f,b2f,b1b,b2b
  __shared__ __bf16   sstage[8][1024];      // per-wave 16x64 bf16 staging (16 KB)

  for (int i = threadIdx.x; i < (32 * 256) / 4; i += 256)
    ((uint4*)sfr)[i] = ((const uint4*)gfrags)[i];
  if (threadIdx.x < 192) sb[threadIdx.x] = gbias[threadIdx.x];
  __syncthreads();

  const int lane = threadIdx.x & 31;
  const int wv   = threadIdx.x >> 5;
  const int half = lane >> 4;
  const int lm   = lane & 15;
  __bf16* stage  = sstage[wv];

  for (int tile = blockIdx.x * 8 + wv; tile < numTiles; tile += gridDim.x * 8) {
    const int eb   = tile << 4;
    const int myE  = eb + lm;
    // Streamed, single-use: non-temporal so they don't evict x/agg from L2.
    const int pastN = __builtin_nontemporal_load(&eidx[myE]);
    const int futN  = __builtin_nontemporal_load(&eidx[NE + myE]);

    // Prefetch next tile's edge_attr rows while we compute this tile.
    __builtin_prefetch(&eattr[(myE + 16) * 16], 0, 1);

    // A fragments: a0 = x[future] (16x32), a1 = x[past], a2 = edge_attr (16x16 pad 32)
    AFrag a0, a1, a2;
#pragma unroll
    for (int j = 0; j < 8; ++j) {
      const int k = a_k(j, half);
      const f2 p = *(const f2*)&x[futN * 32 + k];
      a0.u[j] = pack2(p.x, p.y);
      const f2 q = *(const f2*)&x[pastN * 32 + k];
      a1.u[j] = pack2(q.x, q.y);
    }
#pragma unroll
    for (int j = 0; j < 4; ++j) {
      const int k = (half << 3) + (j << 1);          // K in [0,16)
      const f2 p = __builtin_nontemporal_load((const f2*)&eattr[myE * 16 + k]);
      a2.u[j]     = pack2(p.x, p.y);
      a2.u[j + 4] = 0u;                               // K in [16,32) padded
    }

#pragma unroll
    for (int dir = 0; dir < 2; ++dir) {
      // fwd_in = [x_fut | x_past | attr], bwd_in = [x_past | x_fut | attr]
      const v16bf A0 = dir ? a1.v : a0.v;
      const v16bf A1 = dir ? a0.v : a1.v;
      const unsigned* w1 = &sfr[(dir ? 12 : 0) * 256];
      const unsigned* w2 = &sfr[(24 + dir * 4) * 256];
      const float* b1 = &sb[dir ? 96 : 0];
      const float* b2 = &sb[dir ? 160 : 64];

      // Layer 1: [16x96] @ [96x64] -> relu -> stage (bf16)
#pragma unroll
      for (int nt = 0; nt < 4; ++nt) {
        v8f c = {};
        c = wmma_bf16(A0,   load_frag(w1 + (0 * 4 + nt) * 256 + lane * 8), c);
        c = wmma_bf16(A1,   load_frag(w1 + (1 * 4 + nt) * 256 + lane * 8), c);
        c = wmma_bf16(a2.v, load_frag(w1 + (2 * 4 + nt) * 256 + lane * 8), c);
        const float bb = b1[nt * 16 + lm];
#pragma unroll
        for (int v = 0; v < 8; ++v) {
          float h = c[v] + bb;
          h = h > 0.f ? h : 0.f;
          stage[(v + 8 * half) * 64 + nt * 16 + lm] = (__bf16)h;
        }
      }

      // Transpose C-layout -> A-layout via per-wave LDS staging
      AFrag h0, h1;
#pragma unroll
      for (int j = 0; j < 8; ++j) {
        const int kk = a_k(j, half);
        h0.u[j] = *(const unsigned*)&stage[lm * 64 + kk];
        h1.u[j] = *(const unsigned*)&stage[lm * 64 + 32 + kk];
      }

      // Node indices for the 8 rows this lane's C-fragment covers
      int scat[8];
#pragma unroll
      for (int v = 0; v < 8; ++v) {
        const int row = eb + v + 8 * half;
        scat[v] = dir ? __builtin_nontemporal_load(&eidx[row])
                      : __builtin_nontemporal_load(&eidx[NE + row]);
      }

      // Layer 2: [16x64] @ [64x32] -> relu -> atomic scatter-add
#pragma unroll
      for (int nt = 0; nt < 2; ++nt) {
        v8f c = {};
        c = wmma_bf16(h0.v, load_frag(w2 + (0 * 2 + nt) * 256 + lane * 8), c);
        c = wmma_bf16(h1.v, load_frag(w2 + (1 * 2 + nt) * 256 + lane * 8), c);
        const float bb = b2[nt * 16 + lm];
#pragma unroll
        for (int v = 0; v < 8; ++v) {
          float val = c[v] + bb;
          val = val > 0.f ? val : 0.f;
          atomicAdd(&agg[scat[v] * 64 + dir * 32 + nt * 16 + lm], val);
        }
      }
    }
  }
}

// ---------------------------------------------------------------------------
extern "C" void kernel_launch(void* const* d_in, const int* in_sizes, int n_in,
                              void* d_out, int out_size, void* d_ws, size_t ws_size,
                              hipStream_t stream) {
  const float* x     = (const float*)d_in[0];
  const int*   eidx  = (const int*)d_in[1];     // [2,E] (JAX default int32)
  const float* eattr = (const float*)d_in[2];
  const float* W1f = (const float*)d_in[3];  const float* b1f = (const float*)d_in[4];
  const float* W2f = (const float*)d_in[5];  const float* b2f = (const float*)d_in[6];
  const float* W1b = (const float*)d_in[7];  const float* b1b = (const float*)d_in[8];
  const float* W2b = (const float*)d_in[9];  const float* b2b = (const float*)d_in[10];
  const float* W1n = (const float*)d_in[11]; const float* b1n = (const float*)d_in[12];
  const float* W2n = (const float*)d_in[13]; const float* b2n = (const float*)d_in[14];
  float* out = (float*)d_out;

  // Workspace: agg[N*64] f32, then bf16 fragment table, then bias buffer.
  float*    agg     = (float*)d_ws;
  unsigned* frags   = (unsigned*)((char*)d_ws + AGG_BYTES);
  float*    biasbuf = (float*)((char*)d_ws + AGG_BYTES + FRAG_U32 * 4);

  gnn_zero_kernel<<<1024, 256, 0, stream>>>((float4*)agg, AGG_F32 / 4);
  gnn_pack_kernel<<<(FRAG_U32 + BIAS_F32 + 255) / 256, 256, 0, stream>>>(
      W1f, W2f, W1b, W2b, W1n, W2n, b1f, b2f, b1b, b2b, b1n, b2n, frags, biasbuf);
  gnn_edge_kernel<<<1024, 256, 0, stream>>>(x, eidx, eattr, frags, biasbuf, agg,
                                            NE / 16);
  gnn_node_kernel<<<(NN / 16 + 7) / 8, 256, 0, stream>>>(
      agg, frags + 32 * 256, biasbuf + 192, out, NN / 16);
}
